// ValueNet_55877524521572
// MI455X (gfx1250) — matrix operand order repaired and checked by reading, not compiled, not run on hardware
//
#include <hip/hip_runtime.h>

typedef __attribute__((ext_vector_type(16))) _Float16 v16h;
typedef __attribute__((ext_vector_type(8)))  _Float16 v8h;
typedef __attribute__((ext_vector_type(8)))  float    v8f;
typedef int v4i __attribute__((vector_size(16)));   // pointee type of async builtins

#define N_NODE 64
#define NBATCH 256
#define DN 32
#define DE 16
#define DM 64
#define H_MSG 128
#define H_UPD 256
#define H_OUT 256
#define BN (NBATCH * N_NODE)      /* 16384 rows (b*64+d)   */
#define KUPD (N_NODE * DM)        /* 4096                  */

// ---------------------------------------------------------------------------
// gfx1250 async global->LDS path (ASYNCcnt), with portable fallback
// ---------------------------------------------------------------------------
#if defined(__HIP_DEVICE_COMPILE__) && __has_builtin(__builtin_amdgcn_global_load_async_to_lds_b128)
#define HAVE_ASYNC_LDS 1
#else
#define HAVE_ASYNC_LDS 0
#endif

__device__ __forceinline__ void stage16B(const _Float16* gsrc, _Float16* ldst) {
#if HAVE_ASYNC_LDS
  __builtin_amdgcn_global_load_async_to_lds_b128(
      (__attribute__((address_space(1))) v4i*)(void*)gsrc,
      (__attribute__((address_space(3))) v4i*)(void*)ldst, 0, 0);
#else
  *(v8h*)ldst = *(const v8h*)gsrc;
#endif
}

__device__ __forceinline__ void async_wait0() {
#if defined(__HIP_DEVICE_COMPILE__)
#if HAVE_ASYNC_LDS
#if __has_builtin(__builtin_amdgcn_s_wait_asynccnt)
  __builtin_amdgcn_s_wait_asynccnt(0);
#else
  asm volatile("s_wait_asynccnt 0x0" ::: "memory");
#endif
#endif
#endif
}

// ---------------------------------------------------------------------------
// WMMA fragment helpers (CDNA5 16x16x32 f16, wave32; A/B k order contiguous, not the ISA table; C/D per ISA 7.12.2)
// ---------------------------------------------------------------------------
__device__ __forceinline__ v16h cat8(v8h lo, v8h hi) {
  return __builtin_shufflevector(lo, hi, 0, 1, 2, 3, 4, 5, 6, 7,
                                 8, 9, 10, 11, 12, 13, 14, 15);
}

// A fragment (16x32, f16) from a row-major f16 tile (LDS), stride in halves.
// Lane l: row = l&15; half h -> K = (l>>4)*16 + h
__device__ __forceinline__ v16h load_a_frag(const _Float16* base, int strideH,
                                            int row0, int k0, int lane) {
  const _Float16* p = base + (row0 + (lane & 15)) * strideH + k0 + ((lane >> 4) << 4);
  v8h lo = *(const v8h*)p;
  v8h hi = *(const v8h*)(p + 8);
  return cat8(lo, hi);
}

// B fragment (32x16, f16) from weights pre-transposed to [N][Ktot].
// Lane l: N = l&15; half h -> K = (l>>4)*16 + h  (contiguous 32B in memory)
__device__ __forceinline__ v16h load_b_frag(const _Float16* Bt, int Ktot,
                                            int ncol0, int k0, int lane) {
  const _Float16* p = Bt + (size_t)(ncol0 + (lane & 15)) * Ktot + k0 + ((lane >> 4) << 4);
  return *(const v16h*)p;
}

__device__ __forceinline__ v8f wmma32(v16h a, v16h b, v8f c) {
  return __builtin_amdgcn_wmma_f32_16x16x32_f16(false, a, false, b,
                                                (short)0, c, false, false);
}

// C/D layout: vgpr r, lanes 0-15 -> (M=r, N=lane); lanes 16-31 -> (M=r+8).
__device__ __forceinline__ void store_tile_lds(_Float16* dst, int strideH,
                                               int row0, int col0, int lane,
                                               v8f c, bool relu) {
  int col = col0 + (lane & 15);
  int rb = row0 + ((lane >> 4) << 3);
#pragma unroll
  for (int r = 0; r < 8; ++r) {
    float v = c[r];
    if (relu && v < 0.f) v = 0.f;
    dst[(rb + r) * strideH + col] = (_Float16)v;
  }
}

// ---------------------------------------------------------------------------
// Kernel 0: weight prep  dst[n*Kpad + k] = (f16) src[k*N + n]  (zero-pad K)
// ---------------------------------------------------------------------------
__global__ __launch_bounds__(256) void wprep_kernel(const float* __restrict__ src,
                                                    _Float16* __restrict__ dst,
                                                    int Ksrc, int N, int Kpad) {
  int i = blockIdx.x * 256 + threadIdx.x;
  if (i >= N * Kpad) return;
  int n = i / Kpad, k = i - n * Kpad;
  float v = (k < Ksrc) ? src[(size_t)k * N + n] : 0.f;
  dst[i] = (_Float16)v;
}

// ---------------------------------------------------------------------------
// Kernel 1: per-edge message MLP (48->128->128->64), one WG per (b, s).
// Rows of the tile are d = 0..63; output written transposed into
// msg_t[(b*64+d)*4096 + s*64 + m] so the "scatter" is free.
// ---------------------------------------------------------------------------
#define XS_STR 72    /* 144 B rows: 16B aligned, bank-conflict-skewed */
#define HS_STR 136   /* 272 B rows */

__global__ __launch_bounds__(256) void msg_mlp_kernel(
    const float* __restrict__ nf, const float* __restrict__ ef,
    const _Float16* __restrict__ W1t, const _Float16* __restrict__ W2t,
    const _Float16* __restrict__ W3t,
    const float* __restrict__ b1, const float* __restrict__ b2,
    const float* __restrict__ b3,
    _Float16* __restrict__ msg_t) {
  __shared__ __attribute__((aligned(16))) _Float16 Xs[64 * XS_STR];
  __shared__ __attribute__((aligned(16))) _Float16 H1[64 * HS_STR];
  __shared__ __attribute__((aligned(16))) _Float16 H2[64 * HS_STR];

  const int tid = threadIdx.x;
  const int lane = tid & 31;
  const int w = tid >> 5;
  const int wg = blockIdx.x;          // == b*64 + s == src node index
  const int bidx = wg >> 6;
  const int s = wg & 63;
  const long e0 = (long)wg * 64;      // first edge of this (b,s) block

  // Stage X = [64 rows(d)][64 K]: K<32 src-node feats (same every row),
  // 32<=K<48 edge feats, rest zero-pad.
  for (int i = tid; i < 64 * 64; i += 256) {
    int r = i >> 6, k = i & 63;
    float v = 0.f;
    if (k < DN)            v = nf[(size_t)wg * DN + k];
    else if (k < DN + DE)  v = ef[(e0 + r) * DE + (k - DN)];
    Xs[r * XS_STR + k] = (_Float16)v;
  }
  __syncthreads();

  const int mtile = w & 3;

  // ---- layer 1: [64,64] @ [64,128] + b1, ReLU -> H1 ----
  {
    const int nb = (w >> 2) * 4;
    v8f c[4];
#pragma unroll
    for (int j = 0; j < 4; ++j) {
      float bv = b1[(nb + j) * 16 + (lane & 15)];
#pragma unroll
      for (int r = 0; r < 8; ++r) c[j][r] = bv;
    }
#pragma unroll
    for (int kb = 0; kb < 2; ++kb) {
      v16h bf[4];
#pragma unroll
      for (int j = 0; j < 4; ++j)
        bf[j] = load_b_frag(W1t, 64, (nb + j) * 16, kb * 32, lane);
      v16h a = load_a_frag(Xs, XS_STR, mtile * 16, kb * 32, lane);
#pragma unroll
      for (int j = 0; j < 4; ++j) c[j] = wmma32(a, bf[j], c[j]);
    }
#pragma unroll
    for (int j = 0; j < 4; ++j)
      store_tile_lds(H1, HS_STR, mtile * 16, (nb + j) * 16, lane, c[j], true);
  }
  __syncthreads();

  // ---- layer 2: [64,128] @ [128,128] + b2, ReLU -> H2 ----
  {
    const int nb = (w >> 2) * 4;
    v8f c[4];
#pragma unroll
    for (int j = 0; j < 4; ++j) {
      float bv = b2[(nb + j) * 16 + (lane & 15)];
#pragma unroll
      for (int r = 0; r < 8; ++r) c[j][r] = bv;
    }
#pragma unroll
    for (int kb = 0; kb < 4; ++kb) {
      v16h bf[4];
#pragma unroll
      for (int j = 0; j < 4; ++j)
        bf[j] = load_b_frag(W2t, 128, (nb + j) * 16, kb * 32, lane);
      v16h a = load_a_frag(H1, HS_STR, mtile * 16, kb * 32, lane);
#pragma unroll
      for (int j = 0; j < 4; ++j) c[j] = wmma32(a, bf[j], c[j]);
    }
#pragma unroll
    for (int j = 0; j < 4; ++j)
      store_tile_lds(H2, HS_STR, mtile * 16, (nb + j) * 16, lane, c[j], true);
  }
  __syncthreads();

  // ---- layer 3: [64,128] @ [128,64] + b3 -> msg, transposed global store ----
  {
    const int nb = (w >> 2) * 2;
    v8f c[2];
#pragma unroll
    for (int j = 0; j < 2; ++j) {
      float bv = b3[(nb + j) * 16 + (lane & 15)];
#pragma unroll
      for (int r = 0; r < 8; ++r) c[j][r] = bv;
    }
#pragma unroll
    for (int kb = 0; kb < 4; ++kb) {
      v16h bf[2];
#pragma unroll
      for (int j = 0; j < 2; ++j)
        bf[j] = load_b_frag(W3t, 128, (nb + j) * 16, kb * 32, lane);
      v16h a = load_a_frag(H2, HS_STR, mtile * 16, kb * 32, lane);
#pragma unroll
      for (int j = 0; j < 2; ++j) c[j] = wmma32(a, bf[j], c[j]);
    }
    const int dbase = mtile * 16 + ((lane >> 4) << 3);
#pragma unroll
    for (int j = 0; j < 2; ++j) {
      int col = (nb + j) * 16 + (lane & 15);
#pragma unroll
      for (int r = 0; r < 8; ++r) {
        int d = dbase + r;   // tile row == d
        msg_t[(size_t)(bidx * 64 + d) * KUPD + s * DM + col] = (_Float16)c[j][r];
      }
    }
  }
}

// ---------------------------------------------------------------------------
// Generic GEMM: C[Mtot,NFULL] = act(A[Mtot,K] @ Bt[NFULL,K]^T + bias), f16 io.
// WG = 64 rows; 8 waves = 4 M-tiles x (NFULL/16) N-tiles.
// Double-buffered LDS A tile, staged via async global->LDS (ASYNCcnt);
// all B fragments of a K-step preloaded before the back-to-back wmma chain.
// ---------------------------------------------------------------------------
#define AS_STR 40   /* 80 B rows */

template <int NFULL, bool RELU>
__global__ __launch_bounds__(256) void gemm64_kernel(
    const _Float16* __restrict__ A, const _Float16* __restrict__ Bt,
    const float* __restrict__ bias, _Float16* __restrict__ C, int K) {
  constexpr int NT = NFULL / 32;  // n-tiles per wave
  __shared__ __attribute__((aligned(16))) _Float16 As[2][64 * AS_STR];
  const int tid = threadIdx.x, lane = tid & 31, w = tid >> 5;
  const int mtile = w & 3, nb = (w >> 2) * NT;
  const size_t m0 = (size_t)blockIdx.x * 64;

  v8f c[NT];
#pragma unroll
  for (int j = 0; j < NT; ++j) {
    float bv = bias[(nb + j) * 16 + (lane & 15)];
#pragma unroll
    for (int r = 0; r < 8; ++r) c[j][r] = bv;
  }

  const int srow = tid >> 2, soff = (tid & 3) * 8;
  const _Float16* aptr = A + (m0 + srow) * (size_t)K + soff;
  _Float16* sdst0 = &As[0][srow * AS_STR + soff];
  _Float16* sdst1 = &As[1][srow * AS_STR + soff];

  // prologue: stage K-step 0 into buffer 0
  stage16B(aptr, sdst0);
  async_wait0();
  __syncthreads();

  const int nsteps = K / 32;
  for (int step = 0; step < nsteps; ++step) {
    const int kb = step * 32;
    // stage next tile into the other buffer while we compute this one
    if (step + 1 < nsteps)
      stage16B(aptr + kb + 32, (step & 1) ? sdst0 : sdst1);

    v16h bf[NT];
#pragma unroll
    for (int j = 0; j < NT; ++j)
      bf[j] = load_b_frag(Bt, K, (nb + j) * 16, kb, lane);
    v16h a = load_a_frag(As[step & 1], AS_STR, mtile * 16, 0, lane);
#pragma unroll
    for (int j = 0; j < NT; ++j) c[j] = wmma32(a, bf[j], c[j]);

    async_wait0();     // this wave's async stage landed in LDS
    __syncthreads();   // everyone's stage landed; old buffer reads done
  }

#pragma unroll
  for (int j = 0; j < NT; ++j) {
    int col = (nb + j) * 16 + (lane & 15);
    size_t rb = m0 + mtile * 16 + ((lane >> 4) << 3);
#pragma unroll
    for (int r = 0; r < 8; ++r) {
      float v = c[j][r];
      if (RELU && v < 0.f) v = 0.f;
      C[(rb + r) * NFULL + col] = (_Float16)v;
    }
  }
}

// ---------------------------------------------------------------------------
// Final layer: out[b] = o1[b,:] . out_w2 + out_b2   (N==1, trivial)
// ---------------------------------------------------------------------------
__global__ __launch_bounds__(256) void out_final_kernel(
    const _Float16* __restrict__ o1, const float* __restrict__ w2,
    const float* __restrict__ b2, float* __restrict__ out) {
  int i = threadIdx.x;  // 256 graphs
  float acc = b2[0];
  for (int k = 0; k < H_OUT; ++k) acc += (float)o1[i * H_OUT + k] * w2[k];
  out[i] = acc;
}

// ---------------------------------------------------------------------------
extern "C" void kernel_launch(void* const* d_in, const int* in_sizes, int n_in,
                              void* d_out, int out_size, void* d_ws,
                              size_t ws_size, hipStream_t stream) {
  const float* nf  = (const float*)d_in[0];
  const float* ef  = (const float*)d_in[1];
  // d_in[2] edge_index: all (src,dst) pairs per graph -> structure exploited
  const float* mw1 = (const float*)d_in[3];  const float* mb1 = (const float*)d_in[4];
  const float* mw2 = (const float*)d_in[5];  const float* mb2 = (const float*)d_in[6];
  const float* mw3 = (const float*)d_in[7];  const float* mb3 = (const float*)d_in[8];
  const float* uw1 = (const float*)d_in[9];  const float* ub1 = (const float*)d_in[10];
  const float* uw2 = (const float*)d_in[11]; const float* ub2 = (const float*)d_in[12];
  const float* ow1 = (const float*)d_in[13]; const float* ob1 = (const float*)d_in[14];
  const float* ow2 = (const float*)d_in[15]; const float* ob2 = (const float*)d_in[16];
  float* out = (float*)d_out;

  _Float16* ws = (_Float16*)d_ws;
  size_t off = 0;
  _Float16* W1t  = ws + off; off += 128 * 64;            // msg L1, K padded 48->64
  _Float16* W2t  = ws + off; off += 128 * 128;
  _Float16* W3t  = ws + off; off += 64 * 128;
  _Float16* UW1t = ws + off; off += 256 * (size_t)KUPD;
  _Float16* UW2t = ws + off; off += 64 * 256;
  _Float16* OW1t = ws + off; off += 256 * (size_t)KUPD;
  _Float16* msgT = ws + off; off += (size_t)BN * KUPD;   // [bn, s*64+m]
  _Float16* u1   = ws + off; off += (size_t)BN * 256;
  _Float16* u2   = ws + off; off += (size_t)BN * 64;     // == [256, 4096] view
  _Float16* o1   = ws + off; off += 256 * 256;

  auto g = [](long n) { return (unsigned)((n + 255) / 256); };
  wprep_kernel<<<g(128 * 64), 256, 0, stream>>>(mw1, W1t, 48, 128, 64);
  wprep_kernel<<<g(128 * 128), 256, 0, stream>>>(mw2, W2t, 128, 128, 128);
  wprep_kernel<<<g(64 * 128), 256, 0, stream>>>(mw3, W3t, 128, 64, 128);
  wprep_kernel<<<g(256L * KUPD), 256, 0, stream>>>(uw1, UW1t, KUPD, 256, KUPD);
  wprep_kernel<<<g(64 * 256), 256, 0, stream>>>(uw2, UW2t, 256, 64, 256);
  wprep_kernel<<<g(256L * KUPD), 256, 0, stream>>>(ow1, OW1t, KUPD, 256, KUPD);

  msg_mlp_kernel<<<BN, 256, 0, stream>>>(nf, ef, W1t, W2t, W3t, mb1, mb2, mb3, msgT);
  gemm64_kernel<256, true ><<<BN / 64, 256, 0, stream>>>(msgT, UW1t, ub1, u1, KUPD);
  gemm64_kernel<64,  false><<<BN / 64, 256, 0, stream>>>(u1, UW2t, ub2, u2, 256);
  gemm64_kernel<256, true ><<<4, 256, 0, stream>>>(u2, OW1t, ob1, o1, KUPD);
  out_final_kernel<<<1, 256, 0, stream>>>(o1, ow2, ob2, out);
}
